// ReprogrammingLayer_47734266527807
// MI455X (gfx1250) — matrix operand order repaired and checked by reading, not compiled, not run on hardware
//
#include <hip/hip_runtime.h>
#include <hip/hip_bf16.h>

// ---------------------------------------------------------------------------
// MI455X (gfx1250) implementation of the "reprogramming layer":
//   Q = X@Wq+bq ; K = src@Wk+bk ; scores = Q.K^T per head ;
//   tmps = sum_l softmax_s(scores) ; topk(tmps) ; gather K ; out = tok@Wo+bo
// Big GEMMs use v_wmma_f32_16x16x32_f16 (f16 in, f32 accumulate).
// GEMM waves are register-blocked 32x64 (8 accumulators) so the k-loop runs
// 8 WMMAs per 12 B128 loads instead of 1 per 4.
// ---------------------------------------------------------------------------

typedef __attribute__((ext_vector_type(16))) _Float16 v16h;
typedef __attribute__((ext_vector_type(8)))  _Float16 v8h;
typedef __attribute__((ext_vector_type(8)))  float    v8f;

#define B_DIM 8
#define L_DIM 2048
#define S_DIM 1000
#define D_DIM 1024
#define H_DIM 8
#define E_DIM 128
#define TOPK_N 10
#define S_PAD 1008          // 63 tiles of 16 columns

static __device__ __forceinline__ v16h make_frag(v8h lo, v8h hi) {
  v16h f;
#pragma unroll
  for (int i = 0; i < 8; ++i) { f[i] = lo[i]; f[i + 8] = hi[i]; }
  return f;
}

// --------------------------- prep kernels ----------------------------------

__global__ void cvt_f32_to_f16_kernel(const float* __restrict__ in,
                                      _Float16* __restrict__ out, long n) {
  long i = (long)blockIdx.x * blockDim.x + threadIdx.x;
  long stride = (long)gridDim.x * blockDim.x;
  for (; i < n; i += stride) out[i] = (_Float16)in[i];
}

// out[j][i] = (f16) W[i][j], D=1024 square
__global__ void transpose_cvt_kernel(const float* __restrict__ W,
                                     _Float16* __restrict__ out) {
  long i = (long)blockIdx.x * blockDim.x + threadIdx.x;
  long stride = (long)gridDim.x * blockDim.x;
  const long n = (long)D_DIM * D_DIM;
  for (; i < n; i += stride) {
    int r = (int)(i >> 10), c = (int)(i & 1023);
    out[(long)c * D_DIM + r] = (_Float16)W[i];
  }
}

__global__ void zero_f32_kernel(float* __restrict__ p, long n) {
  long i = (long)blockIdx.x * blockDim.x + threadIdx.x;
  long stride = (long)gridDim.x * blockDim.x;
  for (; i < n; i += stride) p[i] = 0.0f;
}

// --------------------------- WMMA GEMM + bias ------------------------------
// C[M x 1024] = A[M x 1024](f16) * BT[1024 x 1024]^T(f16, stored transposed)
// + bias.  Each wave owns a 32(M) x 64(N) tile: 8 f32 accumulators,
// 2 A-frags x 4 B-frags reused -> 8 wmma per 12 B128 loads per k-step.
// 4 waves per block cover 256 N.  Optionally stores an f32 copy of C.
__global__ __launch_bounds__(128)
void gemm_bias_f16_kernel(const _Float16* __restrict__ A,
                          const _Float16* __restrict__ BT,
                          const float* __restrict__ bias,
                          _Float16* __restrict__ Cf16,
                          float* __restrict__ Cf32, int M) {
  const int lane = threadIdx.x & 31;
  const int wave = threadIdx.x >> 5;
  const int g = lane >> 4;        // lane group (ISA 7.12.2 layouts)
  const int n = lane & 15;

  const int mbase = blockIdx.x * 32;          // 2 m-subtiles of 16
  const int nbase = blockIdx.y * 256 + wave * 64;  // 4 n-subtiles of 16

  // A row pointers for the two m-subtiles (clamped for the M=1000 case)
  const _Float16* ap[2];
#pragma unroll
  for (int ms = 0; ms < 2; ++ms) {
    int arow = mbase + ms * 16 + n;
    if (arow > M - 1) arow = M - 1;
    ap[ms] = A + (size_t)arow * D_DIM;
  }
  // B row pointers for the four n-subtiles
  const _Float16* bp[4];
#pragma unroll
  for (int ns = 0; ns < 4; ++ns)
    bp[ns] = BT + (size_t)(nbase + ns * 16 + n) * D_DIM;

  v8f acc[2][4];
#pragma unroll
  for (int ms = 0; ms < 2; ++ms)
#pragma unroll
    for (int ns = 0; ns < 4; ++ns) acc[ms][ns] = (v8f){};

#pragma unroll 2
  for (int kc = 0; kc < D_DIM / 32; ++kc) {
    v16h af[2];
#pragma unroll
    for (int ms = 0; ms < 2; ++ms) {
      // A frag (16x32): lane holds K = (i&7) + 8g + 16*(i>>3)
      v8h lo = *(const v8h*)(ap[ms] + kc * 32 + 8 * g);
      v8h hi = *(const v8h*)(ap[ms] + kc * 32 + 16 + 8 * g);
      af[ms] = make_frag(lo, hi);
    }
#pragma unroll
    for (int ns = 0; ns < 4; ++ns) {
      // B frag (32x16): lane n = out col, holds K = (i&7) + 8g + 16*(i>>3)
      v8h lo = *(const v8h*)(bp[ns] + kc * 32 + 8 * g);
      v8h hi = *(const v8h*)(bp[ns] + kc * 32 + 16 + 8 * g);
      v16h bf = make_frag(lo, hi);
#pragma unroll
      for (int ms = 0; ms < 2; ++ms)
        acc[ms][ns] = __builtin_amdgcn_wmma_f32_16x16x32_f16(
            false, af[ms], false, bf, (short)0, acc[ms][ns], false, false);
    }
  }

#pragma unroll
  for (int ns = 0; ns < 4; ++ns) {
    const int ncol = nbase + ns * 16 + n;
    const float bv = bias[ncol];
#pragma unroll
    for (int ms = 0; ms < 2; ++ms) {
#pragma unroll
      for (int r = 0; r < 8; ++r) {       // C frag: row = r + 8g, col = n
        int row = mbase + ms * 16 + r + 8 * g;
        if (row < M) {
          float v = acc[ms][ns][r] + bv;
          Cf16[(size_t)row * D_DIM + ncol] = (_Float16)v;
          if (Cf32) Cf32[(size_t)row * D_DIM + ncol] = v;
        }
      }
    }
  }
}

// ------------------- fused scores + softmax + L-reduction ------------------
// grid (L/16, H, B), one wave per block. Computes a 16xS score strip with
// WMMA (Q frags preloaded, reused across all 63 s-tiles), row softmax in
// LDS, then accumulates column sums into tmps[b,h,s] with global atomics.
__global__ __launch_bounds__(32)
void scores_softmax_kernel(const _Float16* __restrict__ Q,
                           const _Float16* __restrict__ K,
                           float* __restrict__ tmps) {
  __shared__ float sc[16 * S_PAD];     // 64512 B
  __shared__ float rowinv[16];
  const int lane = threadIdx.x;
  const int g = lane >> 4;
  const int n = lane & 15;
  const int b = blockIdx.z, h = blockIdx.y;
  const float scale = 0.08838834764831845f;   // 1/sqrt(E)

  const _Float16* qp =
      Q + ((size_t)(b * L_DIM + blockIdx.x * 16 + n)) * D_DIM + h * E_DIM;
  v16h a[4];
#pragma unroll
  for (int ec = 0; ec < 4; ++ec) {
    v8h lo = *(const v8h*)(qp + ec * 32 + 8 * g);
    v8h hi = *(const v8h*)(qp + ec * 32 + 16 + 8 * g);
    a[ec] = make_frag(lo, hi);
  }

  for (int st = 0; st < 63; ++st) {
    const _Float16* kp = K + ((size_t)(st * 16 + n)) * D_DIM + h * E_DIM;
    v8f c = {};
#pragma unroll
    for (int ec = 0; ec < 4; ++ec) {
      v8h lo = *(const v8h*)(kp + ec * 32 + 8 * g);
      v8h hi = *(const v8h*)(kp + ec * 32 + 16 + 8 * g);
      c = __builtin_amdgcn_wmma_f32_16x16x32_f16(false, a[ec],
                                                 false, make_frag(lo, hi),
                                                 (short)0, c, false, false);
    }
    const int col = st * 16 + n;
#pragma unroll
    for (int r = 0; r < 8; ++r) {
      float v = (col < S_DIM) ? scale * c[r] : -__builtin_huge_valf();
      sc[(r + 8 * g) * S_PAD + col] = v;   // pad cols masked to -inf
    }
  }
  __syncthreads();

  { // per-row max + exp + 1/Z ; two lanes per row, each scans half the row
    const int row = lane >> 1, half = lane & 1;
    float* rp = sc + row * S_PAD + half * (S_PAD / 2);
    float m = -__builtin_huge_valf();
    for (int i = 0; i < S_PAD / 2; ++i) m = fmaxf(m, rp[i]);
    m = fmaxf(m, __shfl_xor(m, 1, 32));
    float z = 0.0f;
    for (int i = 0; i < S_PAD / 2; ++i) {
      float e = __expf(rp[i] - m);
      z += e; rp[i] = e;                  // store exp back: reuse below
    }
    z += __shfl_xor(z, 1, 32);
    if (half == 0) rowinv[row] = 1.0f / z;
  }
  __syncthreads();

  float* tp = tmps + (size_t)(b * H_DIM + h) * S_DIM;
  for (int col = lane; col < S_DIM; col += 32) {
    float t = 0.0f;
#pragma unroll
    for (int r = 0; r < 16; ++r) t += sc[r * S_PAD + col] * rowinv[r];
    atomicAdd(tp + col, t);
  }
}

// ------------------------------- top-k -------------------------------------
// One block per (b,h). Iterative argmax (descending order, smallest index
// wins ties -- matches jax.lax.top_k).
__global__ __launch_bounds__(128)
void topk_kernel(const float* __restrict__ tmps, int* __restrict__ idxs) {
  __shared__ float v[S_DIM];
  __shared__ float bv[128];
  __shared__ int   bi[128];
  const int bh = blockIdx.x, tid = threadIdx.x;
  for (int i = tid; i < S_DIM; i += 128) v[i] = tmps[(size_t)bh * S_DIM + i];
  __syncthreads();
  for (int t = 0; t < TOPK_N; ++t) {
    float best = -__builtin_huge_valf(); int besti = -1;
    for (int i = tid; i < S_DIM; i += 128) {
      float x = v[i];
      if (x > best) { best = x; besti = i; }
    }
    bv[tid] = best; bi[tid] = besti;
    __syncthreads();
    if (tid == 0) {
      float Bv = -__builtin_huge_valf(); int Bi = 0;
      for (int j = 0; j < 128; ++j) {
        if (bi[j] < 0) continue;
        if (bv[j] > Bv || (bv[j] == Bv && bi[j] < Bi)) { Bv = bv[j]; Bi = bi[j]; }
      }
      idxs[bh * TOPK_N + t] = Bi;
      v[Bi] = -__builtin_huge_valf();
    }
    __syncthreads();
  }
}

// ---------------------- gather + output GEMM (tiny) ------------------------
// out[(b,t), :] = K_f32[idx(b,h,t), h-slice] @ Wo + bo.  80x1024x1024 FMA.
__global__ __launch_bounds__(256)
void gather_out_kernel(const float* __restrict__ Kf32,
                       const int* __restrict__ idxs,
                       const float* __restrict__ Wo,
                       const float* __restrict__ bo,
                       float* __restrict__ out) {
  const int b = blockIdx.x / TOPK_N, t = blockIdx.x % TOPK_N;
  int rowidx[H_DIM];
#pragma unroll
  for (int h = 0; h < H_DIM; ++h) rowidx[h] = idxs[(b * H_DIM + h) * TOPK_N + t];
  float acc0 = 0.f, acc1 = 0.f, acc2 = 0.f, acc3 = 0.f;
  for (int d = 0; d < D_DIM; ++d) {
    const float kv = Kf32[(size_t)rowidx[d >> 7] * D_DIM + d];
    const float* wr = Wo + (size_t)d * D_DIM + threadIdx.x;
    acc0 += kv * wr[0];   acc1 += kv * wr[256];
    acc2 += kv * wr[512]; acc3 += kv * wr[768];
  }
  float* op = out + (size_t)blockIdx.x * D_DIM + threadIdx.x;
  op[0]   = acc0 + bo[threadIdx.x];
  op[256] = acc1 + bo[threadIdx.x + 256];
  op[512] = acc2 + bo[threadIdx.x + 512];
  op[768] = acc3 + bo[threadIdx.x + 768];
}

// ------------------------------- launch ------------------------------------

extern "C" void kernel_launch(void* const* d_in, const int* in_sizes, int n_in,
                              void* d_out, int out_size, void* d_ws, size_t ws_size,
                              hipStream_t stream) {
  const float* X    = (const float*)d_in[0];   // (B,L,D)
  const float* SRC  = (const float*)d_in[1];   // (S,D)
  const float* Wq   = (const float*)d_in[2];
  const float* bq   = (const float*)d_in[3];
  const float* Wk   = (const float*)d_in[4];
  const float* bk   = (const float*)d_in[5];
  const float* Wo   = (const float*)d_in[6];
  const float* bo   = (const float*)d_in[7];
  float* out = (float*)d_out;

  // workspace carve-up (~80 MB), 256B aligned
  char* ws = (char*)d_ws;
  size_t off = 0;
  auto carve = [&](size_t bytes) {
    char* p = ws + off;
    off = (off + bytes + 255) & ~(size_t)255;
    return p;
  };
  _Float16* Xh    = (_Float16*)carve((size_t)B_DIM * L_DIM * D_DIM * 2);
  _Float16* SRCh  = (_Float16*)carve((size_t)S_DIM * D_DIM * 2);
  _Float16* WqT   = (_Float16*)carve((size_t)D_DIM * D_DIM * 2);
  _Float16* WkT   = (_Float16*)carve((size_t)D_DIM * D_DIM * 2);
  _Float16* Qh    = (_Float16*)carve((size_t)B_DIM * L_DIM * D_DIM * 2);
  _Float16* Kh    = (_Float16*)carve((size_t)S_PAD * D_DIM * 2);
  float*    Kf32  = (float*)   carve((size_t)S_DIM * D_DIM * 4);
  float*    tmps  = (float*)   carve((size_t)B_DIM * H_DIM * S_DIM * 4);
  int*      idxs  = (int*)     carve((size_t)B_DIM * H_DIM * TOPK_N * 4);
  (void)ws_size; (void)n_in; (void)in_sizes; (void)out_size;

  // 1) precision / layout prep
  cvt_f32_to_f16_kernel<<<4096, 256, 0, stream>>>(X, Xh, (long)B_DIM * L_DIM * D_DIM);
  cvt_f32_to_f16_kernel<<<1024, 256, 0, stream>>>(SRC, SRCh, (long)S_DIM * D_DIM);
  transpose_cvt_kernel <<<2048, 256, 0, stream>>>(Wq, WqT);
  transpose_cvt_kernel <<<2048, 256, 0, stream>>>(Wk, WkT);
  zero_f32_kernel      <<<256, 256, 0, stream>>>(tmps, (long)B_DIM * H_DIM * S_DIM);

  // 2) Q = X@Wq+bq (f16 out), K = src@Wk+bk (f16 + f32 copy)
  gemm_bias_f16_kernel<<<dim3(B_DIM * L_DIM / 32, D_DIM / 256), 128, 0, stream>>>(
      Xh, WqT, bq, Qh, nullptr, B_DIM * L_DIM);
  gemm_bias_f16_kernel<<<dim3((S_DIM + 31) / 32, D_DIM / 256), 128, 0, stream>>>(
      SRCh, WkT, bk, Kh, Kf32, S_DIM);

  // 3) fused scores / softmax / L-reduction into tmps
  scores_softmax_kernel<<<dim3(L_DIM / 16, H_DIM, B_DIM), 32, 0, stream>>>(
      Qh, Kh, tmps);

  // 4) top-k per (b,h)
  topk_kernel<<<B_DIM * H_DIM, 128, 0, stream>>>(tmps, idxs);

  // 5) gather + output projection
  gather_out_kernel<<<B_DIM * TOPK_N, 256, 0, stream>>>(Kf32, idxs, Wo, bo, out);
}
